// Interpolator_78683800862819
// MI455X (gfx1250) — matrix operand order repaired and checked by reading, not compiled, not run on hardware
//
#include <hip/hip_runtime.h>
#include <hip/hip_bf16.h>
#include <stdint.h>

typedef __attribute__((ext_vector_type(16))) _Float16 v16h;
typedef __attribute__((ext_vector_type(8)))  _Float16 v8h;
typedef __attribute__((ext_vector_type(8)))  float    v8f;
typedef __attribute__((ext_vector_type(4)))  uint32_t u32x4;
typedef __attribute__((ext_vector_type(8)))  uint32_t u32x8;

#define DCTX 64
#define DYV  32

union AV16 { v16h v; v8h h[2]; };

// ---------------------------------------------------------------------------
// Prep: per row, compute zW, q = z^T W z, and pack f16 U/V rows.
// U[t] = [ztW(64) | zt(64)] ; V[c] = [zc(64) | zcW(64)]  so cross = U V^T (K=128)
// ---------------------------------------------------------------------------
__global__ void interp_prep_rows(const float* __restrict__ zc,
                                 const float* __restrict__ zt,
                                 const float* __restrict__ W,
                                 _Float16* __restrict__ U,
                                 _Float16* __restrict__ V,
                                 float* __restrict__ qtt,
                                 float* __restrict__ qcc,
                                 int T, int C)
{
    int idx = blockIdx.x * blockDim.x + threadIdx.x;
    if (idx >= T + C) return;
    bool isT = idx < T;
    int row  = isT ? idx : idx - T;
    const float* z = (isT ? zt : zc) + (size_t)row * DCTX;

    float zr[DCTX];
#pragma unroll
    for (int k = 0; k < DCTX; ++k) zr[k] = z[k];

    _Float16* dst = (isT ? U : V) + (size_t)row * 128;
    float q = 0.f;
    for (int j = 0; j < DCTX; ++j) {          // W[k*64+j] is wave-uniform -> s_loads
        float acc = 0.f;
#pragma unroll
        for (int k = 0; k < DCTX; ++k) acc += zr[k] * W[k * DCTX + j];
        q += acc * zr[j];
        if (isT) { dst[j] = (_Float16)acc;   dst[64 + j] = (_Float16)zr[j]; }
        else     { dst[j] = (_Float16)zr[j]; dst[64 + j] = (_Float16)acc;  }
    }
    if (isT) qtt[row] = q; else qcc[row] = q;
}

// Yt[dy][c] = y_context[c][dy] in f16 so B-fragments of Y load contiguously.
__global__ void interp_prep_yt(const float* __restrict__ y,
                               _Float16* __restrict__ Yt, int C)
{
    int idx = blockIdx.x * blockDim.x + threadIdx.x;
    if (idx >= DYV * C) return;
    int dy = idx / C, c = idx % C;
    Yt[idx] = (_Float16)y[(size_t)c * DYV + dy];
}

__global__ void interp_zero(float* __restrict__ num, float* __restrict__ den,
                            int nNum, int nDen)
{
    int i = blockIdx.x * blockDim.x + threadIdx.x;
    if (i < nNum) num[i] = 0.f;
    if (i < nDen) den[i] = 0.f;
}

// ---------------------------------------------------------------------------
// Fused main kernel: TDM stages V tiles into double-buffered LDS; scores via
// WMMA -> exp -> theta@Y + theta@1 via WMMA. Block = 256 threads (8 waves);
// wave w owns target rows [m0, m0+16). Grid = (T/128, NSPLIT).
// ---------------------------------------------------------------------------
__global__ __launch_bounds__(256)
void interp_main(const _Float16* __restrict__ U,
                 const _Float16* __restrict__ V,
                 const _Float16* __restrict__ Yt,
                 const float* __restrict__ qtt,
                 const float* __restrict__ qcc,
                 float* __restrict__ num,
                 float* __restrict__ den,
                 int C, int cchunk)
{
    __shared__ __align__(32)  _Float16 lds_th[8][16 * 32];   // per-wave theta tile
    __shared__ __align__(128) _Float16 lds_v[2][32 * 128];   // double-buffered V tile

    const int lane   = threadIdx.x & 31;
    const int w      = threadIdx.x >> 5;
    const int m0     = blockIdx.x * 128 + w * 16;
    const int n      = lane & 15;        // column / row-in-fragment index
    const int hiHalf = lane >> 4;        // 0 or 1

    // ---- TDM: DMA 32x128 f16 tile of V (rows [c0, c0+32)) into lds_v[buf].
    auto issue_tdm = [&](int buf, int c0) {
        uint64_t ga  = (uint64_t)(uintptr_t)(V + (size_t)c0 * 128);
        uint32_t lds = (uint32_t)(uintptr_t)(&lds_v[buf][0]);
        u32x4 g0;
        g0[0] = 1u;                                   // count=1, user descriptor
        g0[1] = lds;                                  // lds_addr [63:32]
        g0[2] = (uint32_t)ga;                         // global_addr [95:64]
        g0[3] = ((uint32_t)(ga >> 32) & 0x01FFFFFFu)  // global_addr [120:96]
                | (2u << 30);                         // type=2 ("image")
        u32x8 g1;
        g1[0] = 0x10000u;                             // wg_mask=0, data_size=1 (2B)
        g1[1] = (128u << 16);                         // tensor_dim0[15:0]=128 at bits63:48
        g1[2] = ((uint32_t)C & 0xFFFFu) << 16;        // tensor_dim1[15:0] at bits95:80
        g1[3] = (((uint32_t)C >> 16) & 0xFFFFu)       // tensor_dim1[31:16]
                | (128u << 16);                       // tile_dim0=128 at bits127:112
        g1[4] = 32u;                                  // tile_dim1=32 (tile_dim2=0)
        g1[5] = 128u;                                 // tensor_dim0_stride[31:0]=128
        g1[6] = 0u;                                   // stride0[47:32]=0, stride1 lo=0
        g1[7] = 0u;
        u32x4 gz = {0u, 0u, 0u, 0u};                  // groups 2/3: zero (2-D tile)
        asm volatile("tensor_load_to_lds %0, %1, %2, %3"
                     :: "s"(g0), "s"(g1), "s"(gz), "s"(gz)
                     : "memory");
    };

    // --- A fragments of U (16 rows x K=128 as 4 slabs of 32), contiguous k order (same in B):
    //     half h of lane l holds K = 32f + 16*(l>=16) + h
    AV16 aU[4];
    {
        const _Float16* ur = U + (size_t)(m0 + n) * 128 + hiHalf * 16;
#pragma unroll
        for (int f = 0; f < 4; ++f) {
            aU[f].h[0] = *(const v8h*)(ur + f * 32);
            aU[f].h[1] = *(const v8h*)(ur + f * 32 + 8);
        }
    }

    // q_tt for this wave's rows in C/D layout (VGPR r, lane half -> M = 8*hi + r)
    float qt[8];
#pragma unroll
    for (int r = 0; r < 8; ++r) qt[r] = qtt[m0 + hiHalf * 8 + r];

    v16h onesv;
#pragma unroll
    for (int i = 0; i < 16; ++i) onesv[i] = (_Float16)1.0f;

    v8f accO0 = {}, accO1 = {}, accD = {};

    const int cBeg = blockIdx.y * cchunk;
    const int cEnd = cBeg + cchunk;
    _Float16* th = &lds_th[w][0];

    // prologue: fetch first tile
    if (w == 0) {
        issue_tdm(0, cBeg);
        __builtin_amdgcn_s_wait_tensorcnt(0);
    }
    __syncthreads();

    int buf = 0;
    for (int c0 = cBeg; c0 < cEnd; c0 += 32) {
        const bool more = (c0 + 32) < cEnd;
        if (w == 0 && more) issue_tdm(buf ^ 1, c0 + 32);   // overlap DMA with compute

        // ---- batch-load ALL 8 B fragments first (one DS clause, one wait),
        //      then run the 8-WMMA chain back-to-back.
        const _Float16* vr0 = &lds_v[buf][(size_t)n * 128 + hiHalf * 16];       // B col n
        const _Float16* vr1 = vr0 + 16 * 128;                                   // cols +16
        AV16 b[8];
#pragma unroll
        for (int f = 0; f < 4; ++f) {
            b[f].h[0]     = *(const v8h*)(vr0 + f * 32);
            b[f].h[1]     = *(const v8h*)(vr0 + f * 32 + 8);
            b[4 + f].h[0] = *(const v8h*)(vr1 + f * 32);
            b[4 + f].h[1] = *(const v8h*)(vr1 + f * 32 + 8);
        }

        v8f S0 = {}, S1 = {};
#pragma unroll
        for (int f = 0; f < 4; ++f) {
            S0 = __builtin_amdgcn_wmma_f32_16x16x32_f16(false, aU[f].v, false, b[f].v,
                                                        (short)0, S0, false, false);
            S1 = __builtin_amdgcn_wmma_f32_16x16x32_f16(false, aU[f].v, false, b[4 + f].v,
                                                        (short)0, S1, false, false);
        }

        // ---- theta = exp(-(qtt + qcc - cross)) = exp2((cross - qtt - qcc)*log2e)
        const float qc0 = qcc[c0 + n];
        const float qc1 = qcc[c0 + 16 + n];
#pragma unroll
        for (int r = 0; r < 8; ++r) {
            float t0 = __builtin_amdgcn_exp2f((S0[r] - qt[r] - qc0) * 1.44269504088896f);
            float t1 = __builtin_amdgcn_exp2f((S1[r] - qt[r] - qc1) * 1.44269504088896f);
            th[(hiHalf * 8 + r) * 32 + n]      = (_Float16)t0;   // row-major 16x32
            th[(hiHalf * 8 + r) * 32 + 16 + n] = (_Float16)t1;
        }

        // ---- reload theta as A fragment (16 rows x K=32 contexts) and load Y
        //      B-fragments before the accumulate-WMMA chain.
        AV16 thA, y0, y1;
        thA.h[0] = *(const v8h*)(th + n * 32 + hiHalf * 16);
        thA.h[1] = *(const v8h*)(th + n * 32 + 8 + hiHalf * 16);
        const _Float16* yr0 = Yt + (size_t)n * C + c0 + hiHalf * 16;
        const _Float16* yr1 = Yt + (size_t)(16 + n) * C + c0 + hiHalf * 16;
        y0.h[0] = *(const v8h*)(yr0); y0.h[1] = *(const v8h*)(yr0 + 8);
        y1.h[0] = *(const v8h*)(yr1); y1.h[1] = *(const v8h*)(yr1 + 8);

        accO0 = __builtin_amdgcn_wmma_f32_16x16x32_f16(false, thA.v, false, y0.v,
                                                       (short)0, accO0, false, false);
        accO1 = __builtin_amdgcn_wmma_f32_16x16x32_f16(false, thA.v, false, y1.v,
                                                       (short)0, accO1, false, false);
        accD  = __builtin_amdgcn_wmma_f32_16x16x32_f16(false, thA.v, false, onesv,
                                                       (short)0, accD, false, false);

        if (w == 0 && more) __builtin_amdgcn_s_wait_tensorcnt(0);
        __syncthreads();          // next buffer ready; all reads of cur done
        buf ^= 1;
    }

    // ---- combine split partials
#pragma unroll
    for (int r = 0; r < 8; ++r) {
        int m = m0 + hiHalf * 8 + r;
        atomicAdd(&num[(size_t)m * DYV + n],      accO0[r]);
        atomicAdd(&num[(size_t)m * DYV + 16 + n], accO1[r]);
    }
    if (n == 0) {
#pragma unroll
        for (int r = 0; r < 8; ++r)
            atomicAdd(&den[m0 + hiHalf * 8 + r], accD[r]);  // every col of accD = row sum
    }
}

__global__ void interp_finalize(const float* __restrict__ num,
                                const float* __restrict__ den,
                                float* __restrict__ out, int nTot)
{
    int i = blockIdx.x * blockDim.x + threadIdx.x;
    if (i < nTot) out[i] = num[i] / den[i / DYV];
}

// ---------------------------------------------------------------------------
extern "C" void kernel_launch(void* const* d_in, const int* in_sizes, int n_in,
                              void* d_out, int out_size, void* d_ws, size_t ws_size,
                              hipStream_t stream)
{
    const float* zc = (const float*)d_in[0];   // (C, 64)
    const float* yc = (const float*)d_in[1];   // (C, 32)
    const float* zt = (const float*)d_in[2];   // (T, 64)
    const float* W  = (const float*)d_in[3];   // (64, 64)
    const int C = in_sizes[0] / DCTX;
    const int T = in_sizes[2] / DCTX;
    float* out = (float*)d_out;

    // Workspace carve (~5.6 MB total)
    uintptr_t p = (uintptr_t)d_ws;
    auto alignup = [&](uintptr_t a) { p = (p + a - 1) & ~(a - 1); };
    alignup(256);
    _Float16* U  = (_Float16*)p; p += (size_t)T * 128 * sizeof(_Float16);
    _Float16* V  = (_Float16*)p; p += (size_t)C * 128 * sizeof(_Float16);
    alignup(256);
    _Float16* Yt = (_Float16*)p; p += (size_t)DYV * C * sizeof(_Float16);
    alignup(256);
    float* qtt = (float*)p; p += (size_t)T * sizeof(float);
    float* qcc = (float*)p; p += (size_t)C * sizeof(float);
    float* num = (float*)p; p += (size_t)T * DYV * sizeof(float);
    float* den = (float*)p; p += (size_t)T * sizeof(float);

    interp_prep_rows<<<(T + C + 255) / 256, 256, 0, stream>>>(zc, zt, W, U, V, qtt, qcc, T, C);
    interp_prep_yt<<<(DYV * C + 255) / 256, 256, 0, stream>>>(yc, Yt, C);
    interp_zero<<<(T * (DYV + 1) + 255) / 256, 256, 0, stream>>>(num, den, T * DYV, T);

    const int NSPLIT = 8;                 // 64 row-tiles x 8 C-splits = 512 blocks
    const int cchunk = C / NSPLIT;
    interp_main<<<dim3(T / 128, NSPLIT), 256, 0, stream>>>(U, V, Yt, qtt, qcc, num, den, C, cchunk);

    interp_finalize<<<(T * DYV + 255) / 256, 256, 0, stream>>>(num, den, out, T * DYV);
}
